// Enhance_58703613002034
// MI455X (gfx1250) — compile-verified, matrix operand order checked by reading, not run on hardware
//
#include <hip/hip_runtime.h>
#include <hip/hip_bf16.h>

typedef __attribute__((ext_vector_type(16))) __bf16 v16bf;
typedef __attribute__((ext_vector_type(8)))  float  v8f;
typedef __attribute__((ext_vector_type(8)))  __bf16 v8bf;

#define C_DIM 512
#define D_DIM 256
#define HW    4096
#define B_DIM 16
#define NROWS 65536   // B*HW
#define INV_SQRT_C 0.04419417382415922f

// ---------------------------------------------------------------------------
// WMMA helpers: A stored (M,K) K-contiguous, B-source stored (N,K) K-contiguous.
// 16-bit fragment layout (ISA 7.12.2): lanes 0-15 -> K {kb..kb+7, kb+16..kb+23},
// lanes 16-31 -> K {kb+8..kb+15, kb+24..kb+31}; row/col = lane&15.
// ---------------------------------------------------------------------------
__device__ __forceinline__ v16bf frag_ld(const __bf16* __restrict__ base, int ld,
                                         int r0, int kb, int lane) {
  int r  = r0 + (lane & 15);
  int ko = kb + ((lane & 16) ? 8 : 0);
  const __bf16* p = base + (size_t)r * ld + ko;
  union { v16bf v; v8bf h[2]; } u;
  u.h[0] = *(const v8bf*)(p);
  u.h[1] = *(const v8bf*)(p + 16);
  return u.v;
}

__device__ __forceinline__ void wmma_core(const __bf16* __restrict__ A, int lda,
                                          const __bf16* __restrict__ B, int ldb,
                                          int m0, int n0, int K, int lane,
                                          v8f acc[2][2]) {
  for (int kb = 0; kb < K; kb += 32) {
    v16bf a0 = frag_ld(A, lda, m0,      kb, lane);
    v16bf a1 = frag_ld(A, lda, m0 + 16, kb, lane);
    v16bf b0 = frag_ld(B, ldb, n0,      kb, lane);
    v16bf b1 = frag_ld(B, ldb, n0 + 16, kb, lane);
    acc[0][0] = __builtin_amdgcn_wmma_f32_16x16x32_bf16(false, a0, false, b0, (short)0, acc[0][0], false, false);
    acc[0][1] = __builtin_amdgcn_wmma_f32_16x16x32_bf16(false, a0, false, b1, (short)0, acc[0][1], false, false);
    acc[1][0] = __builtin_amdgcn_wmma_f32_16x16x32_bf16(false, a1, false, b0, (short)0, acc[1][0], false, false);
    acc[1][1] = __builtin_amdgcn_wmma_f32_16x16x32_bf16(false, a1, false, b1, (short)0, acc[1][1], false, false);
  }
}

// GEMM1: Y[n,o] = sum_c X[n,c] * W[o,c]  (both K-contiguous), store bf16
__global__ void k_gemm_store_bf16(const __bf16* __restrict__ A,
                                  const __bf16* __restrict__ B,
                                  __bf16* __restrict__ Y, int K, int N) {
  int lane = threadIdx.x & 31, wid = threadIdx.x >> 5;
  int m0 = blockIdx.x * 128 + (wid & 3) * 32;
  int n0 = blockIdx.y * 64  + (wid >> 2) * 32;
  v8f acc[2][2] = {};
  wmma_core(A, K, B, K, m0, n0, K, lane, acc);
  int colb = lane & 15;
  int roff = (lane & 16) ? 8 : 0;
  for (int i = 0; i < 2; ++i)
    for (int j = 0; j < 2; ++j) {
      int col = n0 + j * 16 + colb;
      int rb  = m0 + i * 16 + roff;
      #pragma unroll
      for (int e = 0; e < 8; ++e)
        Y[(size_t)(rb + e) * N + col] = (__bf16)acc[i][j][e];
    }
}

// logits: L[n,d] = scale * sum_c A[n,c] * Qt[d,c], store f32
__global__ void k_gemm_logits(const __bf16* __restrict__ A,
                              const __bf16* __restrict__ B,
                              float* __restrict__ L, int K, int N, float scale) {
  int lane = threadIdx.x & 31, wid = threadIdx.x >> 5;
  int m0 = blockIdx.x * 128 + (wid & 3) * 32;
  int n0 = blockIdx.y * 64  + (wid >> 2) * 32;
  v8f acc[2][2] = {};
  wmma_core(A, K, B, K, m0, n0, K, lane, acc);
  int colb = lane & 15;
  int roff = (lane & 16) ? 8 : 0;
  for (int i = 0; i < 2; ++i)
    for (int j = 0; j < 2; ++j) {
      int col = n0 + j * 16 + colb;
      int rb  = m0 + i * 16 + roff;
      #pragma unroll
      for (int e = 0; e < 8; ++e)
        L[(size_t)(rb + e) * N + col] = acc[i][j][e] * scale;
    }
}

// GEMM2 + epilogue: out[b,c,n] = (P @ Q^T)[n,c] * gsig[b,c] + x[b,c,n]
__global__ void k_gemm_out(const __bf16* __restrict__ P,
                           const __bf16* __restrict__ Qb,
                           const float* __restrict__ gsig,
                           const float* __restrict__ x,
                           float* __restrict__ out) {
  int lane = threadIdx.x & 31, wid = threadIdx.x >> 5;
  int m0 = blockIdx.x * 128 + (wid & 3) * 32;
  int n0 = blockIdx.y * 64  + (wid >> 2) * 32;
  v8f acc[2][2] = {};
  wmma_core(P, D_DIM, Qb, D_DIM, m0, n0, D_DIM, lane, acc);
  int b = (blockIdx.x * 128) >> 12;   // 128 | 4096, so no straddle
  int colb = lane & 15;
  int roff = (lane & 16) ? 8 : 0;
  for (int i = 0; i < 2; ++i)
    for (int j = 0; j < 2; ++j) {
      int col = n0 + j * 16 + colb;                // channel c
      int rb  = m0 + i * 16 + roff;                // global row n
      float gate = gsig[b * C_DIM + col];
      size_t base = ((size_t)(b * C_DIM + col)) * HW + (rb & (HW - 1));
      #pragma unroll
      for (int e = 0; e < 8; ++e)
        out[base + e] = acc[i][j][e] * gate + x[base + e];
    }
}

// ---------------------------------------------------------------------------
// QR via Modified Gram-Schmidt, in place, one workgroup (512 threads, 16 waves)
// ---------------------------------------------------------------------------
__global__ void k_qr(const float* __restrict__ W, float* __restrict__ V) {
  int t = threadIdx.x;                 // 0..511 == row
  for (int i = t; i < C_DIM * D_DIM; i += C_DIM) V[i] = W[i] + 1e-8f;
  __syncthreads();
  int lane = t & 31, wid = t >> 5;
  __shared__ float red[C_DIM];
  __shared__ float s_rinv;
  for (int j = 0; j < D_DIM; ++j) {
    float v = V[t * D_DIM + j];
    red[t] = v * v;
    __syncthreads();
    for (int s = 256; s > 0; s >>= 1) { if (t < s) red[t] += red[t + s]; __syncthreads(); }
    if (t == 0) s_rinv = rsqrtf(red[0] + 1e-20f);
    __syncthreads();
    float q = v * s_rinv;
    V[t * D_DIM + j] = q;
    __syncthreads();
    for (int k = j + 1 + wid; k < D_DIM; k += 16) {
      float dot = 0.f;
      for (int r = lane; r < C_DIM; r += 32) dot += V[r * D_DIM + j] * V[r * D_DIM + k];
      for (int m = 16; m; m >>= 1) dot += __shfl_xor(dot, m, 32);
      for (int r = lane; r < C_DIM; r += 32) V[r * D_DIM + k] -= dot * V[r * D_DIM + j];
    }
    __syncthreads();
  }
}

// ---------------------------------------------------------------------------
// Small / elementwise kernels
// ---------------------------------------------------------------------------
__global__ void k_cvt_bf16(const float* __restrict__ s, __bf16* __restrict__ d, int n) {
  int i = blockIdx.x * blockDim.x + threadIdx.x;
  for (; i < n; i += gridDim.x * blockDim.x) d[i] = (__bf16)s[i];
}

__global__ void k_qcvt(const float* __restrict__ Q, __bf16* __restrict__ Qb,
                       __bf16* __restrict__ Qt) {
  int i = blockIdx.x * 256 + threadIdx.x;      // 131072 elems
  float v = Q[i];
  Qb[i] = (__bf16)v;
  int c = i >> 8, d = i & 255;
  Qt[d * C_DIM + c] = (__bf16)v;
}

// x (b,c,hw) f32 -> xt (b,hw,c) bf16 via LDS tiles
__global__ void k_xt(const float* __restrict__ x, __bf16* __restrict__ xt) {
  __shared__ float tile[32][33];
  int b = blockIdx.z, n0 = blockIdx.x * 32, c0 = blockIdx.y * 32;
  const float* xb = x + (size_t)b * C_DIM * HW;
  for (int i = threadIdx.y; i < 32; i += 8)
    tile[i][threadIdx.x] = xb[(size_t)(c0 + i) * HW + n0 + threadIdx.x];
  __syncthreads();
  __bf16* xo = xt + (size_t)b * HW * C_DIM;
  for (int i = threadIdx.y; i < 32; i += 8)
    xo[(size_t)(n0 + i) * C_DIM + c0 + threadIdx.x] = (__bf16)tile[threadIdx.x][i];
}

// per-channel sum / sumsq over (b,hw) rows of Y (NROWS x 512 bf16)
__global__ void k_stats(const __bf16* __restrict__ Y, float* __restrict__ sum,
                        float* __restrict__ sq) {
  int t = threadIdx.x;                         // 256
  size_t r0 = (size_t)blockIdx.x * 256;        // 256 blocks
  float s0 = 0, q0 = 0, s1 = 0, q1 = 0;
  for (int r = 0; r < 256; ++r) {
    const __bf16* row = Y + (r0 + r) * C_DIM;
    float a = (float)row[t];
    float b = (float)row[t + 256];
    s0 += a; q0 += a * a; s1 += b; q1 += b * b;
  }
  atomicAdd(&sum[t], s0);       atomicAdd(&sq[t], q0);
  atomicAdd(&sum[t + 256], s1); atomicAdd(&sq[t + 256], q1);
}

__global__ void k_scales(const float* __restrict__ sum, const float* __restrict__ sq,
                         const float* __restrict__ gamma, const float* __restrict__ beta,
                         float* __restrict__ a, float* __restrict__ bsh) {
  int c = threadIdx.x;                         // 512
  const float N = (float)NROWS;
  float m = sum[c] / N;
  float v = sq[c] / N - m * m;
  float s = gamma[c] * rsqrtf(v + 1e-5f);
  a[c] = s; bsh[c] = beta[c] - m * s;
}

// g[b,o] = mean_n relu(a*gx + b)
__global__ void k_pool(const __bf16* __restrict__ Y, const float* __restrict__ a,
                       const float* __restrict__ bsh, float* __restrict__ g) {
  int b = blockIdx.x, o = threadIdx.x;         // 16 x 512
  float sa = a[o], sb = bsh[o], acc = 0.f;
  const __bf16* p = Y + (size_t)b * HW * C_DIM + o;
  for (int n = 0; n < HW; ++n) acc += fmaxf(sa * (float)p[(size_t)n * C_DIM] + sb, 0.f);
  g[b * C_DIM + o] = acc * (1.f / (float)HW);
}

__global__ void k_bnrelu(__bf16* __restrict__ Y, const float* __restrict__ a,
                         const float* __restrict__ bsh) {
  size_t i = (size_t)blockIdx.x * blockDim.x + threadIdx.x;
  size_t n = (size_t)NROWS * C_DIM;
  size_t step = (size_t)gridDim.x * blockDim.x;
  for (; i < n; i += step) {
    int c = (int)(i & (C_DIM - 1));
    Y[i] = (__bf16)fmaxf(a[c] * (float)Y[i] + bsh[c], 0.f);
  }
}

// aff_g = softmax_d( (g @ Q) / sqrt(c) )
__global__ void k_affg(const float* __restrict__ g, const float* __restrict__ Q,
                       float* __restrict__ affg) {
  int b = blockIdx.x, d = threadIdx.x;         // 16 x 256
  const float* gb = g + b * C_DIM;
  float l = 0.f;
  for (int c = 0; c < C_DIM; ++c) l += gb[c] * Q[c * D_DIM + d];
  l *= INV_SQRT_C;
  __shared__ float red[D_DIM];
  red[d] = l; __syncthreads();
  for (int s = 128; s; s >>= 1) { if (d < s) red[d] = fmaxf(red[d], red[d + s]); __syncthreads(); }
  float mx = red[0]; __syncthreads();
  float e = __expf(l - mx);
  red[d] = e; __syncthreads();
  for (int s = 128; s; s >>= 1) { if (d < s) red[d] += red[d + s]; __syncthreads(); }
  affg[b * D_DIM + d] = e / red[0];
}

// gsig = sigmoid(aff_g @ Q^T)
__global__ void k_gsig(const float* __restrict__ affg, const float* __restrict__ Q,
                       float* __restrict__ gsig) {
  int b = blockIdx.x, c = threadIdx.x;         // 16 x 512
  const float* ab = affg + b * D_DIM;
  const float* qr = Q + c * D_DIM;
  float s = 0.f;
  for (int d = 0; d < D_DIM; ++d) s += ab[d] * qr[d];
  gsig[b * C_DIM + c] = 1.f / (1.f + __expf(-s));
}

// row softmax (256 wide), one wave per row, write bf16 P
__global__ void k_softmax(const float* __restrict__ L, __bf16* __restrict__ P) {
  int lane = threadIdx.x & 31, wid = threadIdx.x >> 5;
  size_t row = (size_t)blockIdx.x * 8 + wid;
  const float* lr = L + row * D_DIM;
  float v[8], mx = -3.4e38f;
  #pragma unroll
  for (int i = 0; i < 8; ++i) { v[i] = lr[lane + 32 * i]; mx = fmaxf(mx, v[i]); }
  for (int m = 16; m; m >>= 1) mx = fmaxf(mx, __shfl_xor(mx, m, 32));
  float s = 0.f;
  #pragma unroll
  for (int i = 0; i < 8; ++i) { v[i] = __expf(v[i] - mx); s += v[i]; }
  for (int m = 16; m; m >>= 1) s += __shfl_xor(s, m, 32);
  float inv = 1.f / s;
  __bf16* pr = P + row * D_DIM;
  #pragma unroll
  for (int i = 0; i < 8; ++i) pr[lane + 32 * i] = (__bf16)(v[i] * inv);
}

// ---------------------------------------------------------------------------
extern "C" void kernel_launch(void* const* d_in, const int* in_sizes, int n_in,
                              void* d_out, int out_size, void* d_ws, size_t ws_size,
                              hipStream_t stream) {
  (void)in_sizes; (void)n_in; (void)out_size; (void)ws_size;
  const float* x    = (const float*)d_in[0];
  const float* wgl  = (const float*)d_in[1];
  const float* Wg   = (const float*)d_in[2];
  const float* ggam = (const float*)d_in[3];
  const float* gbet = (const float*)d_in[4];
  const float* Wc   = (const float*)d_in[5];
  const float* cgam = (const float*)d_in[6];
  const float* cbet = (const float*)d_in[7];
  float* out = (float*)d_out;

  char* ws = (char*)d_ws;
  auto alloc = [&](size_t sz) { void* p = (void*)ws; ws += (sz + 255) & ~(size_t)255; return p; };
  float*  Qf  = (float*) alloc((size_t)C_DIM * D_DIM * 4);
  __bf16* Qb  = (__bf16*)alloc((size_t)C_DIM * D_DIM * 2);
  __bf16* Qt  = (__bf16*)alloc((size_t)C_DIM * D_DIM * 2);
  __bf16* Wgb = (__bf16*)alloc((size_t)C_DIM * C_DIM * 2);
  __bf16* Wcb = (__bf16*)alloc((size_t)C_DIM * C_DIM * 2);
  __bf16* xt  = (__bf16*)alloc((size_t)NROWS * C_DIM * 2);
  __bf16* Y   = (__bf16*)alloc((size_t)NROWS * C_DIM * 2);
  float*  Lg  = (float*) alloc((size_t)NROWS * D_DIM * 4);
  __bf16* P   = (__bf16*)alloc((size_t)NROWS * D_DIM * 2);
  float* ssum = (float*) alloc(C_DIM * 4);
  float* ssq  = (float*) alloc(C_DIM * 4);
  float* asc  = (float*) alloc(C_DIM * 4);
  float* bsh  = (float*) alloc(C_DIM * 4);
  float* g    = (float*) alloc(B_DIM * C_DIM * 4);
  float* affg = (float*) alloc(B_DIM * D_DIM * 4);
  float* gsig = (float*) alloc(B_DIM * C_DIM * 4);

  // 1) QR (in place -> Qf) + bf16 copies of Q / Q^T / conv weights
  k_qr<<<1, 512, 0, stream>>>(wgl, Qf);
  k_qcvt<<<(C_DIM * D_DIM) / 256, 256, 0, stream>>>(Qf, Qb, Qt);
  k_cvt_bf16<<<256, 256, 0, stream>>>(Wg, Wgb, C_DIM * C_DIM);
  k_cvt_bf16<<<256, 256, 0, stream>>>(Wc, Wcb, C_DIM * C_DIM);

  // 2) x -> (b, hw, c) bf16
  k_xt<<<dim3(HW / 32, C_DIM / 32, B_DIM), dim3(32, 8), 0, stream>>>(x, xt);

  // 3) global branch: conv GEMM -> BN stats -> fused BN+ReLU+pool -> gate
  k_gemm_store_bf16<<<dim3(NROWS / 128, C_DIM / 64), 256, 0, stream>>>(xt, Wgb, Y, C_DIM, C_DIM);
  hipMemsetAsync(ssum, 0, C_DIM * 4, stream);
  hipMemsetAsync(ssq,  0, C_DIM * 4, stream);
  k_stats<<<NROWS / 256, 256, 0, stream>>>(Y, ssum, ssq);
  k_scales<<<1, C_DIM, 0, stream>>>(ssum, ssq, ggam, gbet, asc, bsh);
  k_pool<<<B_DIM, C_DIM, 0, stream>>>(Y, asc, bsh, g);
  k_affg<<<B_DIM, D_DIM, 0, stream>>>(g, Qf, affg);
  k_gsig<<<B_DIM, C_DIM, 0, stream>>>(affg, Qf, gsig);

  // 4) channel branch: conv GEMM (reuse Y) -> BN stats -> BN+ReLU in place
  k_gemm_store_bf16<<<dim3(NROWS / 128, C_DIM / 64), 256, 0, stream>>>(xt, Wcb, Y, C_DIM, C_DIM);
  hipMemsetAsync(ssum, 0, C_DIM * 4, stream);
  hipMemsetAsync(ssq,  0, C_DIM * 4, stream);
  k_stats<<<NROWS / 256, 256, 0, stream>>>(Y, ssum, ssq);
  k_scales<<<1, C_DIM, 0, stream>>>(ssum, ssq, cgam, cbet, asc, bsh);
  k_bnrelu<<<8192, 256, 0, stream>>>(Y, asc, bsh);

  // 5) logits GEMM -> row softmax -> P @ Q^T with gated residual epilogue
  k_gemm_logits<<<dim3(NROWS / 128, D_DIM / 64), 256, 0, stream>>>(Y, Qt, Lg, C_DIM, D_DIM, INV_SQRT_C);
  k_softmax<<<NROWS / 8, 256, 0, stream>>>(Lg, P);
  k_gemm_out<<<dim3(NROWS / 128, C_DIM / 64), 256, 0, stream>>>(P, Qb, gsig, x, out);
}
